// ChannelBlock_38053410243417
// MI455X (gfx1250) — matrix operand order repaired and checked by reading, not compiled, not run on hardware
//
#include <hip/hip_runtime.h>
#include <hip/hip_bf16.h>
#include <math.h>

// ---------------------------------------------------------------------------
// CDNA5 / gfx1250: all GEMMs on v_wmma_f32_16x16x32_bf16, A/B tiles staged in
// LDS by the Tensor Data Mover (tensor_load_to_lds, TENSORcnt), double
// buffered. Weights pre-transposed to [N][K] so every tile is a plain 2D
// row-major 128x32 bf16 block (perfect TDM descriptor fit, no LDS scatter).
// ---------------------------------------------------------------------------

typedef __bf16 bf16_t;
typedef __bf16 v16bf __attribute__((ext_vector_type(16)));
typedef float  v8f   __attribute__((ext_vector_type(8)));
typedef unsigned int v4u __attribute__((ext_vector_type(4)));
typedef int v4i __attribute__((ext_vector_type(4)));
typedef int v8i __attribute__((ext_vector_type(8)));

union FragU { v16bf v; int4 q[2]; };

#define NTOK   32768
#define DIMC   256
#define HIDDEN 1024
#define NQKV   768
#define HEADS  8
#define HD     32
#define SEQ    4096

// ---------------------------------------------------------------------------
// TDM: async-load a 128-row x 32-col bf16 tile (row stride = k_total elements)
// from global into LDS at byte offset lds_off. Descriptor per ISA 08 §8:
//   group0: count=1 | lds_addr | global_addr(57b) | type=2
//   group1: data_size=1(2B), tensor_dim0=k_total, tensor_dim1=rows_total,
//           tile_dim0=32, tile_dim1=128, tensor_dim0_stride=k_total
// All inputs are workgroup-uniform -> descriptor lives in SGPRs.
// ---------------------------------------------------------------------------
__device__ __forceinline__ void tdm_load_tile_128x32(unsigned lds_off,
                                                     const bf16_t* gptr,
                                                     unsigned k_total,
                                                     unsigned rows_total) {
    unsigned long long ga = (unsigned long long)(uintptr_t)gptr;
    v4u g0;
    g0[0] = 1u;                                           // count=1 (valid), user mode
    g0[1] = lds_off;                                      // lds_addr (bytes)
    g0[2] = (unsigned)(ga & 0xFFFFFFFFu);                 // global_addr[31:0]
    g0[3] = (unsigned)((ga >> 32) & 0x01FFFFFFu)          // global_addr[56:32]
            | (2u << 30);                                 // type=2 ("image")
    v8i g1;
    g1[0] = 0x00010000;                                   // data_size=1 (2 bytes)
    g1[1] = (int)((k_total & 0xFFFFu) << 16);             // tensor_dim0[15:0] @ [63:48]
    g1[2] = (int)((k_total >> 16) | ((rows_total & 0xFFFFu) << 16)); // dim0 hi | dim1 lo
    g1[3] = (int)((rows_total >> 16) | (32u << 16));      // dim1 hi | tile_dim0=32
    g1[4] = 128;                                          // tile_dim1=128, tile_dim2=0
    g1[5] = (int)k_total;                                 // tensor_dim0_stride[31:0]
    g1[6] = 0;                                            // stride hi, dim1_stride lo
    g1[7] = 0;
    v4i z4 = {0, 0, 0, 0};
#if __clang_major__ >= 23
    v8i z8 = {0, 0, 0, 0, 0, 0, 0, 0};
    __builtin_amdgcn_tensor_load_to_lds(g0, g1, z4, z4, z8, 0);
#else
    __builtin_amdgcn_tensor_load_to_lds(g0, g1, z4, z4, 0);
#endif
}

// ---------------------------------------------------------------------------
// fp32 [K][N] weight  ->  bf16 [N][K] transposed weight
// block 256 = 32 k-lanes x 8 n-waves ; grid (K/32, N/8)
// ---------------------------------------------------------------------------
__global__ __launch_bounds__(256) void transpose_w_to_bf16(const float* __restrict__ in,
                                                           bf16_t* __restrict__ out,
                                                           int K, int N) {
    int k = blockIdx.x * 32 + (threadIdx.x & 31);
    int n = blockIdx.y * 8 + (threadIdx.x >> 5);
    out[(size_t)n * K + k] = (bf16_t)in[(size_t)k * N + n];
}

// ---------------------------------------------------------------------------
// LayerNorm over C=256, one wave per row, bf16 out. grid rows/8, block 256.
// ---------------------------------------------------------------------------
__global__ __launch_bounds__(256) void layernorm_to_bf16(const float* __restrict__ x,
                                                         const float* __restrict__ gam,
                                                         const float* __restrict__ bet,
                                                         bf16_t* __restrict__ out) {
    const int w    = threadIdx.x >> 5;
    const int lane = threadIdx.x & 31;
    const int row  = blockIdx.x * 8 + w;
    const float* xr = x + (size_t)row * DIMC;

    float vals[8];
    float s = 0.f;
#pragma unroll
    for (int j = 0; j < 8; ++j) { vals[j] = xr[lane + j * 32]; s += vals[j]; }
#pragma unroll
    for (int m = 16; m >= 1; m >>= 1) s += __shfl_xor(s, m, 32);
    const float mu = s * (1.0f / DIMC);

    float vs = 0.f;
#pragma unroll
    for (int j = 0; j < 8; ++j) { float d = vals[j] - mu; vs += d * d; }
#pragma unroll
    for (int m = 16; m >= 1; m >>= 1) vs += __shfl_xor(vs, m, 32);
    const float inv = rsqrtf(vs * (1.0f / DIMC) + 1e-5f);

    bf16_t* orow = out + (size_t)row * DIMC;
#pragma unroll
    for (int j = 0; j < 8; ++j) {
        int c = lane + j * 32;
        orow[c] = (bf16_t)((vals[j] - mu) * inv * gam[c] + bet[c]);
    }
}

// ---------------------------------------------------------------------------
// WMMA GEMM: C[M,N] = A[M,K] * Bt[N,K]^T (+epilogue). Tile 128x128x32,
// 256 threads = 8 waves (2x4), each wave 4x2 16x16 accumulators.
// A and B tiles are both 128x32 bf16 row-major blocks, TDM double-buffered:
//   LDS: A0 @0, A1 @8K, B0 @16K, B1 @24K  (dynamic LDS, 32 KB)
// Wave 0 issues tensor_load_to_lds for slab i+1 while all waves compute
// slab i, then s_wait_tensorcnt 0 + workgroup barrier flips buffers.
// EPI: 0 = +bias -> bf16 ; 1 = +bias+res1 -> f32 ;
//      2 = gelu(+bias) -> bf16 ; 3 = +bias+res1+res2 -> f32
// ---------------------------------------------------------------------------
template <int EPI>
__global__ __launch_bounds__(256) void gemm_bf16_wmma(const bf16_t* __restrict__ A,
                                                      const bf16_t* __restrict__ Bt,
                                                      const float* __restrict__ bias,
                                                      const float* __restrict__ res1,
                                                      const float* __restrict__ res2,
                                                      bf16_t* __restrict__ outb,
                                                      float* __restrict__ outf,
                                                      int M, int N, int K) {
    extern __shared__ __align__(16) char smem[];   // 32 KB dynamic

    const int tid  = threadIdx.x;
    const int lane = tid & 31;
    const int w    = tid >> 5;     // wave 0..7
    const int wm   = w >> 2;       // 0..1  (64 rows each)
    const int wn   = w & 3;        // 0..3  (32 cols each)
    const int g    = lane >> 4;
    const int r    = lane & 15;
    const int blockM = blockIdx.x * 128;
    const int blockN = blockIdx.y * 128;

    const bf16_t* Ablk = A  + (size_t)blockM * K;
    const bf16_t* Bblk = Bt + (size_t)blockN * K;

    v8f acc[4][2];
#pragma unroll
    for (int mi = 0; mi < 4; ++mi)
#pragma unroll
        for (int ni = 0; ni < 2; ++ni)
#pragma unroll
            for (int e = 0; e < 8; ++e) acc[mi][ni][e] = 0.f;

    // prologue: fetch slab 0 into buffer 0
    if (tid < 32) {
        tdm_load_tile_128x32(0u,     Ablk, (unsigned)K, (unsigned)M);
        tdm_load_tile_128x32(16384u, Bblk, (unsigned)K, (unsigned)N);
        __builtin_amdgcn_s_wait_tensorcnt(0);
    }
    __syncthreads();

    int buf = 0;
    for (int k0 = 0; k0 < K; k0 += 32) {
        // issue next slab's DMA into the other buffer (overlaps with compute)
        if (k0 + 32 < K && tid < 32) {
            tdm_load_tile_128x32((unsigned)((buf ^ 1) * 8192),         Ablk + (k0 + 32),
                                 (unsigned)K, (unsigned)M);
            tdm_load_tile_128x32((unsigned)(16384 + (buf ^ 1) * 8192), Bblk + (k0 + 32),
                                 (unsigned)K, (unsigned)N);
        }

        const bf16_t (*As)[32] = (const bf16_t(*)[32])(smem + buf * 8192);
        const bf16_t (*Bs)[32] = (const bf16_t(*)[32])(smem + 16384 + buf * 8192);

        FragU afr[4], bfr[2];
#pragma unroll
        for (int mi = 0; mi < 4; ++mi) {
            int row = wm * 64 + mi * 16 + r;
            afr[mi].q[0] = *(const int4*)(&As[row][8 * g]);
            afr[mi].q[1] = *(const int4*)(&As[row][16 + 8 * g]);
        }
#pragma unroll
        for (int ni = 0; ni < 2; ++ni) {
            int col = wn * 32 + ni * 16 + r;
            bfr[ni].q[0] = *(const int4*)(&Bs[col][8 * g]);
            bfr[ni].q[1] = *(const int4*)(&Bs[col][16 + 8 * g]);
        }
#pragma unroll
        for (int mi = 0; mi < 4; ++mi)
#pragma unroll
            for (int ni = 0; ni < 2; ++ni)
                acc[mi][ni] = __builtin_amdgcn_wmma_f32_16x16x32_bf16(
                    false, afr[mi].v, false, bfr[ni].v, (short)0, acc[mi][ni], false, false);

        if (tid < 32) __builtin_amdgcn_s_wait_tensorcnt(0);
        __syncthreads();
        buf ^= 1;
    }

    // epilogue (C layout: col = r, row = vr + 8*g within each 16x16 tile)
#pragma unroll
    for (int mi = 0; mi < 4; ++mi) {
#pragma unroll
        for (int ni = 0; ni < 2; ++ni) {
            const int colg = blockN + wn * 32 + ni * 16 + r;
            const float bb = bias[colg];
#pragma unroll
            for (int vr = 0; vr < 8; ++vr) {
                const int rowg = blockM + wm * 64 + mi * 16 + vr + 8 * g;
                const size_t idx = (size_t)rowg * N + colg;
                float v = acc[mi][ni][vr] + bb;
                if constexpr (EPI == 0) {
                    outb[idx] = (bf16_t)v;
                } else if constexpr (EPI == 1) {
                    outf[idx] = v + res1[idx];
                } else if constexpr (EPI == 2) {
                    float t = 0.5f * v * (1.0f + erff(v * 0.70710678118654752f));
                    outb[idx] = (bf16_t)t;
                } else {
                    outf[idx] = v + res1[idx] + res2[idx];
                }
            }
        }
    }
}

// ---------------------------------------------------------------------------
// Channel-attention stats: per (b,h), S[d][e] = sum_n k[n,d]*v[n,e],
// softmax over e with scale hd^-0.5. Stores attn[d][e] row-major bf16.
// grid=64, block=128 (4 waves split N=4096; wave-private LDS regions).
// ---------------------------------------------------------------------------
__global__ __launch_bounds__(128) void attn_stats(const bf16_t* __restrict__ qkv,
                                                  bf16_t* __restrict__ attnA) {
    const int bh = blockIdx.x;
    const int b  = bh >> 3;
    const int h  = bh & 7;
    const int tid  = threadIdx.x;
    const int lane = tid & 31;
    const int w    = tid >> 5;
    const int g    = lane >> 4;
    const int r    = lane & 15;

    __shared__ __align__(16) unsigned char pool[16384];
    bf16_t (*kT)[32] = (bf16_t(*)[32])(pool + (size_t)w * 4096);         // [d][n]
    bf16_t (*vT)[32] = (bf16_t(*)[32])(pool + (size_t)w * 4096 + 2048);  // [e][n]
    float* red = (float*)pool;

    v8f acc[2][2];
#pragma unroll
    for (int mi = 0; mi < 2; ++mi)
#pragma unroll
        for (int ni = 0; ni < 2; ++ni)
#pragma unroll
            for (int e = 0; e < 8; ++e) acc[mi][ni][e] = 0.f;

    for (int s = w; s < SEQ / 32; s += 4) {
        const int tok = b * SEQ + s * 32 + lane;
        const bf16_t* kp = qkv + (size_t)tok * NQKV + DIMC + h * HD;
        const bf16_t* vp = kp + DIMC;
        int4 kt[4], vt[4];
#pragma unroll
        for (int q = 0; q < 4; ++q) kt[q] = ((const int4*)kp)[q];
#pragma unroll
        for (int q = 0; q < 4; ++q) vt[q] = ((const int4*)vp)[q];
        const bf16_t* ke = (const bf16_t*)kt;
        const bf16_t* ve = (const bf16_t*)vt;
#pragma unroll
        for (int d = 0; d < 32; ++d) kT[d][lane] = ke[d];
#pragma unroll
        for (int d = 0; d < 32; ++d) vT[d][lane] = ve[d];

        FragU afr[2], bfr[2];
#pragma unroll
        for (int mi = 0; mi < 2; ++mi) {
            afr[mi].q[0] = *(const int4*)(&kT[mi * 16 + r][8 * g]);
            afr[mi].q[1] = *(const int4*)(&kT[mi * 16 + r][16 + 8 * g]);
        }
#pragma unroll
        for (int ni = 0; ni < 2; ++ni) {
            bfr[ni].q[0] = *(const int4*)(&vT[ni * 16 + r][8 * g]);
            bfr[ni].q[1] = *(const int4*)(&vT[ni * 16 + r][16 + 8 * g]);
        }
#pragma unroll
        for (int mi = 0; mi < 2; ++mi)
#pragma unroll
            for (int ni = 0; ni < 2; ++ni)
                acc[mi][ni] = __builtin_amdgcn_wmma_f32_16x16x32_bf16(
                    false, afr[mi].v, false, bfr[ni].v, (short)0, acc[mi][ni], false, false);
    }

    // per-wave partial 32x32 into own region (overlays kT/vT)
#pragma unroll
    for (int mi = 0; mi < 2; ++mi)
#pragma unroll
        for (int ni = 0; ni < 2; ++ni)
#pragma unroll
            for (int vr = 0; vr < 8; ++vr)
                red[w * 1024 + (mi * 16 + vr + 8 * g) * 32 + ni * 16 + r] = acc[mi][ni][vr];
    __syncthreads();

#pragma unroll
    for (int i = 0; i < 8; ++i) {
        int idx = tid * 8 + i;
        red[idx] = red[idx] + red[1024 + idx] + red[2048 + idx] + red[3072 + idx];
    }
    __syncthreads();

    if (tid < 32) {
        const float scale = 0.17677669529663687f;  // 1/sqrt(32)
        const int d = tid;
        float mx = -1e30f;
#pragma unroll
        for (int e = 0; e < 32; ++e) mx = fmaxf(mx, red[d * 32 + e] * scale);
        float p[32], sum = 0.f;
#pragma unroll
        for (int e = 0; e < 32; ++e) { p[e] = __expf(red[d * 32 + e] * scale - mx); sum += p[e]; }
        const float inv = 1.0f / sum;
        bf16_t* out = attnA + (size_t)bh * 1024 + d * 32;
#pragma unroll
        for (int e = 0; e < 32; ++e) out[e] = (bf16_t)(p[e] * inv);
    }
}

// ---------------------------------------------------------------------------
// Apply: y[n, h*32+d] = sum_e q[n, h*32+e] * attn[d][e]  (K=32, 1 WMMA step)
// grid NTOK/64, block 256; wave w = head w; fragments direct from global.
// ---------------------------------------------------------------------------
__global__ __launch_bounds__(256) void attn_apply(const bf16_t* __restrict__ qkv,
                                                  const bf16_t* __restrict__ attnA,
                                                  bf16_t* __restrict__ y) {
    const int t0   = blockIdx.x * 64;
    const int b    = t0 >> 12;
    const int tid  = threadIdx.x;
    const int lane = tid & 31;
    const int w    = tid >> 5;
    const int g    = lane >> 4;
    const int r    = lane & 15;
    const int bh   = b * HEADS + w;

    FragU bfr[2];
#pragma unroll
    for (int ni = 0; ni < 2; ++ni) {
        const bf16_t* p = attnA + (size_t)bh * 1024 + (ni * 16 + r) * 32 + 8 * g;
        bfr[ni].q[0] = ((const int4*)p)[0];
        bfr[ni].q[1] = ((const int4*)p)[2];
    }

    v8f acc[4][2];
#pragma unroll
    for (int mi = 0; mi < 4; ++mi)
#pragma unroll
        for (int ni = 0; ni < 2; ++ni)
#pragma unroll
            for (int e = 0; e < 8; ++e) acc[mi][ni][e] = 0.f;

#pragma unroll
    for (int mi = 0; mi < 4; ++mi) {
        const int tok = t0 + mi * 16 + r;
        const bf16_t* qp = qkv + (size_t)tok * NQKV + w * HD;
        FragU afr;
        afr.q[0] = *(const int4*)(qp + 8 * g);
        afr.q[1] = *(const int4*)(qp + 16 + 8 * g);
#pragma unroll
        for (int ni = 0; ni < 2; ++ni)
            acc[mi][ni] = __builtin_amdgcn_wmma_f32_16x16x32_bf16(
                false, afr.v, false, bfr[ni].v, (short)0, acc[mi][ni], false, false);
    }

#pragma unroll
    for (int mi = 0; mi < 4; ++mi)
#pragma unroll
        for (int ni = 0; ni < 2; ++ni) {
            const int colg = w * HD + ni * 16 + r;
#pragma unroll
            for (int vr = 0; vr < 8; ++vr) {
                const int rowg = t0 + mi * 16 + vr + 8 * g;
                y[(size_t)rowg * DIMC + colg] = (bf16_t)acc[mi][ni][vr];
            }
        }
}

// ---------------------------------------------------------------------------
// Host-side launch
// ---------------------------------------------------------------------------
extern "C" void kernel_launch(void* const* d_in, const int* in_sizes, int n_in,
                              void* d_out, int out_size, void* d_ws, size_t ws_size,
                              hipStream_t stream) {
    const float* x      = (const float*)d_in[0];
    const float* qkv_w  = (const float*)d_in[1];
    const float* qkv_b  = (const float*)d_in[2];
    const float* proj_w = (const float*)d_in[3];
    const float* proj_b = (const float*)d_in[4];
    const float* ln1_g  = (const float*)d_in[5];
    const float* ln1_b  = (const float*)d_in[6];
    const float* ln2_g  = (const float*)d_in[7];
    const float* ln2_b  = (const float*)d_in[8];
    const float* fc1_w  = (const float*)d_in[9];
    const float* fc1_b  = (const float*)d_in[10];
    const float* fc2_w  = (const float*)d_in[11];
    const float* fc2_b  = (const float*)d_in[12];
    float* out = (float*)d_out;

    char* ws = (char*)d_ws;
    size_t off = 0;
    auto alloc = [&](size_t bytes) { size_t o = off; off += (bytes + 255) & ~(size_t)255; return o; };
    const size_t o_wqkv = alloc((size_t)DIMC * NQKV * 2);    // [768][256] bf16
    const size_t o_wprj = alloc((size_t)DIMC * DIMC * 2);    // [256][256]
    const size_t o_wfc1 = alloc((size_t)DIMC * HIDDEN * 2);  // [1024][256]
    const size_t o_wfc2 = alloc((size_t)HIDDEN * DIMC * 2);  // [256][1024]
    const size_t o_xln  = alloc((size_t)NTOK * DIMC * 2);    // LN1 out, reused for LN2
    const size_t o_big  = alloc((size_t)NTOK * HIDDEN * 2);  // qkv (48MB), then gelu acts
    const size_t o_y    = alloc((size_t)NTOK * DIMC * 2);
    const size_t o_attn = alloc((size_t)64 * 32 * 32 * 2);
    const size_t o_h    = alloc((size_t)NTOK * DIMC * 4);
    (void)ws_size; (void)n_in; (void)in_sizes; (void)out_size;

    bf16_t* wqkvT = (bf16_t*)(ws + o_wqkv);
    bf16_t* wprjT = (bf16_t*)(ws + o_wprj);
    bf16_t* wfc1T = (bf16_t*)(ws + o_wfc1);
    bf16_t* wfc2T = (bf16_t*)(ws + o_wfc2);
    bf16_t* xln   = (bf16_t*)(ws + o_xln);
    bf16_t* qkv   = (bf16_t*)(ws + o_big);
    bf16_t* gact  = (bf16_t*)(ws + o_big);    // aliases dead qkv buffer
    bf16_t* ybuf  = (bf16_t*)(ws + o_y);
    bf16_t* attnA = (bf16_t*)(ws + o_attn);
    float*  hbuf  = (float*)(ws + o_h);

    // 1) weights -> bf16, transposed to [N][K]
    transpose_w_to_bf16<<<dim3(DIMC / 32, NQKV / 8),  256, 0, stream>>>(qkv_w,  wqkvT, DIMC,   NQKV);
    transpose_w_to_bf16<<<dim3(DIMC / 32, DIMC / 8),  256, 0, stream>>>(proj_w, wprjT, DIMC,   DIMC);
    transpose_w_to_bf16<<<dim3(DIMC / 32, HIDDEN / 8),256, 0, stream>>>(fc1_w,  wfc1T, DIMC,   HIDDEN);
    transpose_w_to_bf16<<<dim3(HIDDEN / 32, DIMC / 8),256, 0, stream>>>(fc2_w,  wfc2T, HIDDEN, DIMC);

    // 2) LN1
    layernorm_to_bf16<<<NTOK / 8, 256, 0, stream>>>(x, ln1_g, ln1_b, xln);

    // 3) QKV GEMM (+bias) -> bf16
    gemm_bf16_wmma<0><<<dim3(NTOK / 128, NQKV / 128), 256, 32768, stream>>>(
        xln, wqkvT, qkv_b, nullptr, nullptr, qkv, nullptr, NTOK, NQKV, DIMC);

    // 4) channel-attention stats + softmax
    attn_stats<<<64, 128, 0, stream>>>(qkv, attnA);

    // 5) apply attention -> y
    attn_apply<<<NTOK / 64, 256, 0, stream>>>(qkv, attnA, ybuf);

    // 6) proj GEMM + bias + residual(x) -> h (f32)
    gemm_bf16_wmma<1><<<dim3(NTOK / 128, DIMC / 128), 256, 32768, stream>>>(
        ybuf, wprjT, proj_b, x, nullptr, nullptr, hbuf, NTOK, DIMC, DIMC);

    // 7) LN2 (reuse xln buffer)
    layernorm_to_bf16<<<NTOK / 8, 256, 0, stream>>>(hbuf, ln2_g, ln2_b, xln);

    // 8) FC1 + bias + exact GELU -> bf16
    gemm_bf16_wmma<2><<<dim3(NTOK / 128, HIDDEN / 128), 256, 32768, stream>>>(
        xln, wfc1T, fc1_b, nullptr, nullptr, gact, nullptr, NTOK, HIDDEN, DIMC);

    // 9) FC2 + bias + h + shortcut(x) -> out (f32)
    gemm_bf16_wmma<3><<<dim3(NTOK / 128, DIMC / 128), 256, 32768, stream>>>(
        gact, wfc2T, fc2_b, hbuf, x, nullptr, out, NTOK, DIMC, HIDDEN);
}
